// Classification_Head_57346403336763
// MI455X (gfx1250) — matrix operand order repaired and checked by reading, not compiled, not run on hardware
//
#include <hip/hip_runtime.h>

#define BATCH 8
#define SEQ   1900
#define LP    1920          // padded length (multiple of 32; 1920 = 15*128)
#define EMB   256
#define NH    8
#define HD    32
#define QT_PER (LP/16)      // 120 query tiles per (b,h)
#define KSTEPS (LP/32)      // 60 key steps of 32
#define QBLK  (LP/128)      // 15 query slabs of 128 per (b,h)

typedef _Float16 half_t;
typedef __attribute__((ext_vector_type(16))) _Float16 v16h;
typedef __attribute__((ext_vector_type(8)))  _Float16 v8h;
typedef __attribute__((ext_vector_type(8)))  float    v8f;
typedef __attribute__((ext_vector_type(4)))  int      i32x4;

#if defined(__has_builtin)
#if __has_builtin(__builtin_amdgcn_global_load_async_to_lds_b128) && \
    __has_builtin(__builtin_amdgcn_s_wait_asynccnt)
#define HAVE_ASYNC_LDS 1
#endif
#endif

__device__ __forceinline__ v8f wmma_f16(v16h a, v16h b, v8f c) {
  // D = A(16x32 f16) * B(32x16 f16) + C(16x16 f32)
  return __builtin_amdgcn_wmma_f32_16x16x32_f16(
      /*neg_a=*/false, a, /*neg_b=*/false, b,
      /*c_mod=*/(short)0, c, /*reuse_a=*/false, /*reuse_b=*/false);
}

// Build A-operand (16x32, M=lane&15, K=hh*16+j contiguous) from two
// contiguous 8-half chunks at [hh*16] and [hh*16+8] of the row.
__device__ __forceinline__ v16h load_a16x32(const half_t* row_base, int hh) {
  const half_t* ap = row_base + hh * 16;
  v8h a0 = *(const v8h*)ap;
  v8h a1 = *(const v8h*)(ap + 8);
  v16h a;
#pragma unroll
  for (int j = 0; j < 8; ++j) { a[j] = a0[j]; a[8 + j] = a1[j]; }
  return a;
}

// Stage one 16-byte chunk global->LDS (async DMA if available).
__device__ __forceinline__ void stage_b128(const half_t* gsrc, half_t* ldst) {
#ifdef HAVE_ASYNC_LDS
  __builtin_amdgcn_global_load_async_to_lds_b128(
      (__attribute__((address_space(1))) i32x4*)gsrc,
      (__attribute__((address_space(3))) i32x4*)ldst, 0, 0);
#else
  *(v8h*)ldst = *(const v8h*)gsrc;
#endif
}

__device__ __forceinline__ void stage_wait() {
#ifdef HAVE_ASYNC_LDS
  __builtin_amdgcn_s_wait_asynccnt(0);
#endif
}

// ---------------------------------------------------------------- prep ------
__global__ void prep_x_kernel(const float* __restrict__ x, half_t* __restrict__ xh) {
  int i   = blockIdx.x * 256 + threadIdx.x;    // over BATCH*LP*EMB
  int e   = i & (EMB - 1);
  int rem = i >> 8;                            // b*LP + l
  int l   = rem % LP;
  int b   = rem / LP;
  float v = 0.0f;
  if (l < SEQ) v = x[((size_t)(b * SEQ + l)) * EMB + e];
  xh[i] = (half_t)v;
}

__global__ void prep_w_kernel(const float* __restrict__ wqkv, const float* __restrict__ wout,
                              half_t* __restrict__ wqkvh, half_t* __restrict__ wouth) {
  int i = blockIdx.x * 256 + threadIdx.x;      // over 4*EMB*EMB
  if (i < 3 * EMB * EMB) wqkvh[i] = (half_t)wqkv[i];
  else                   wouth[i - 3 * EMB * EMB] = (half_t)wout[i - 3 * EMB * EMB];
}

// ------------------------------------------------------------ QKV GEMM -----
__global__ void qkv_gemm_kernel(const half_t* __restrict__ xh, const half_t* __restrict__ wh,
                                const float* __restrict__ bias,
                                half_t* __restrict__ Qh, half_t* __restrict__ Kh,
                                half_t* __restrict__ Vt) {
  const int NT = 3 * EMB / 16;                 // 48 column tiles
  int wid  = threadIdx.x >> 5;
  int lane = threadIdx.x & 31;
  int tile = blockIdx.x * 8 + wid;
  int mt = tile / NT, nt = tile % NT;
  int n = lane & 15, hh = lane >> 4;           // n doubles as A-row index m

  const half_t* arow = xh + (size_t)(mt * 16 + n) * EMB;
  const half_t* brow = wh + (size_t)(nt * 16 + n) * EMB;

  v8f acc = {};
#pragma unroll
  for (int ks = 0; ks < EMB / 32; ++ks) {
    v16h a  = load_a16x32(arow + ks * 32, hh);
    v16h bm = *(const v16h*)(brow + ks * 32 + hh * 16);  // B: K=hh*16+j contiguous
    acc = wmma_f16(a, bm, acc);
  }

  int f  = nt * 16 + n;
  float bv = bias[f];
  int h = (f >> 5) & (NH - 1);
  int d = f & 31;
#pragma unroll
  for (int r = 0; r < 8; ++r) {
    int row = mt * 16 + r + 8 * hh;            // C layout: row = r + 8*(lane/16)
    int b = row / LP, l = row % LP;
    half_t val = (half_t)(acc[r] + bv);
    if (f < EMB)            Qh[((size_t)(b * NH + h) * LP + l) * HD + d] = val;
    else if (f < 2 * EMB)   Kh[((size_t)(b * NH + h) * LP + l) * HD + d] = val;
    else                    Vt[((size_t)(b * NH + h) * HD + d) * LP + l] = val;
  }
}

// --------------------------------------------------------- flash attention --
// One block = one (b,h) and a 128-query slab (8 waves x 16 queries). K/V
// 32-key tiles are staged into LDS once per step and shared by all 8 waves.
__global__ void attn_kernel(const half_t* __restrict__ Qh, const half_t* __restrict__ Kh,
                            const half_t* __restrict__ Vt,
                            const int* __restrict__ pad_size_p,
                            const int* __restrict__ single_pad_p,
                            half_t* __restrict__ ctxh) {
  __shared__ half_t ldsK[32 * 32];             // [local_key][d]
  __shared__ half_t ldsV[32 * 32];             // [d][local_key]
  __shared__ half_t plds[8][16 * 32];          // per-wave P staging (C->A relayout)

  int wid  = threadIdx.x >> 5;
  int lane = threadIdx.x & 31;
  int bh = blockIdx.x / QBLK;
  int qb = blockIdx.x % QBLK;
  int qt = qb * 8 + wid;
  int b  = bh / NH;
  int h  = bh % NH;
  int n = lane & 15, hh = lane >> 4;

  const int pad_size = *pad_size_p;
  const int dn2      = 2 * (*single_pad_p);
  const float scale  = 0.17677669529663687f;   // 1/sqrt(32)

  const half_t* Qp = Qh + (size_t)bh * LP * HD;
  const half_t* Kp = Kh + (size_t)bh * LP * HD;
  const half_t* Vp = Vt + (size_t)bh * HD * LP;

  // Per-thread staging assignment: threads 0-127 stage K, 128-255 stage V,
  // one 16B chunk each (32 rows x 4 chunks).
  int st = threadIdx.x;
  int srow = (st & 127) >> 2;
  int scol = (st & 3) * 8;

  // Q A-tile: K-dim == D == 32, loaded once.
  v16h qa = load_a16x32(Qp + (size_t)(qt * 16 + n) * HD, hh);

  v8f acc0 = {}, acc1 = {};                    // 16x32 context accumulator (f32)
  float mrow[8], lrow[8];
#pragma unroll
  for (int r = 0; r < 8; ++r) { mrow[r] = -1e30f; lrow[r] = 0.0f; }

  half_t* myp = plds[wid];

  for (int ks = 0; ks < KSTEPS; ++ks) {
    int key0 = ks * 32;

    __syncthreads();                           // previous-step LDS reads done
    if (st < 128)
      stage_b128(Kp + (size_t)(key0 + srow) * HD + scol, ldsK + srow * 32 + scol);
    else
      stage_b128(Vp + (size_t)srow * LP + key0 + scol, ldsV + srow * 32 + scol);
    stage_wait();                              // our async transfers landed
    __syncthreads();                           // tiles visible to all waves

    // S = Q*K^T : B[k=d][n=key] from LDS K tile
    v16h kb0 = *(const v16h*)(ldsK + n * 32 + hh * 16);
    v16h kb1 = *(const v16h*)(ldsK + (16 + n) * 32 + hh * 16);
    v8f z = {};
    v8f sa = wmma_f16(qa, kb0, z);
    v8f sb = wmma_f16(qa, kb1, z);

    int kj0 = key0 + n,      kj1 = key0 + 16 + n;
    bool kin0 = kj0 < pad_size, kin1 = kj1 < pad_size;
    bool kpad0 = kj0 >= SEQ,    kpad1 = kj1 >= SEQ;
    int kg0 = kj0 / dn2,        kg1 = kj1 / dn2;

#pragma unroll
    for (int r = 0; r < 8; ++r) {
      int ql  = qt * 16 + r + 8 * hh;
      bool qin = ql < pad_size;
      int  qg  = ql / dn2;
      float s0 = sa[r] * scale, s1 = sb[r] * scale;
      // DN mask: key masked if padded, or (DN key && (non-DN query || other group))
      if (kpad0 || (kin0 && (!qin || qg != kg0))) s0 = -1e30f;
      if (kpad1 || (kin1 && (!qin || qg != kg1))) s1 = -1e30f;

      float tm = fmaxf(s0, s1);
#pragma unroll
      for (int off = 1; off < 16; off <<= 1) tm = fmaxf(tm, __shfl_xor(tm, off, 16));
      float mnew  = fmaxf(mrow[r], tm);
      float alpha = __expf(mrow[r] - mnew);    // underflows to 0 when old=-1e30
      float p0 = __expf(s0 - mnew);
      float p1 = __expf(s1 - mnew);
      float rs = p0 + p1;
#pragma unroll
      for (int off = 1; off < 16; off <<= 1) rs += __shfl_xor(rs, off, 16);
      lrow[r] = lrow[r] * alpha + rs;
      mrow[r] = mnew;
      acc0[r] *= alpha;
      acc1[r] *= alpha;
      // stage P row-major 16x32 in LDS (wave-internal, DS ops are in-order)
      myp[(r + 8 * hh) * 32 + n]      = (half_t)p0;
      myp[(r + 8 * hh) * 32 + 16 + n] = (half_t)p1;
    }

    // reload P in A layout, ctx += P * V  (V tile [d][key] -> contiguous reads)
    v16h pa  = load_a16x32(myp + n * 32, hh);
    v16h vb0 = *(const v16h*)(ldsV + n * 32 + hh * 16);
    v16h vb1 = *(const v16h*)(ldsV + (16 + n) * 32 + hh * 16);
    acc0 = wmma_f16(pa, vb0, acc0);
    acc1 = wmma_f16(pa, vb1, acc1);
  }

#pragma unroll
  for (int r = 0; r < 8; ++r) {
    int ql = qt * 16 + r + 8 * hh;
    float inv = 1.0f / fmaxf(lrow[r], 1e-20f);
    size_t base = ((size_t)(b * LP + ql)) * EMB + h * HD;
    ctxh[base + n]      = (half_t)(acc0[r] * inv);
    ctxh[base + 16 + n] = (half_t)(acc1[r] * inv);
  }
}

// ---------------------------------------------------------- out projection --
__global__ void outproj_gemm_kernel(const half_t* __restrict__ ctxh,
                                    const half_t* __restrict__ wouth,
                                    const float* __restrict__ ob,
                                    float* __restrict__ attn_out) {
  const int NT = EMB / 16;                     // 16
  int wid  = threadIdx.x >> 5;
  int lane = threadIdx.x & 31;
  int tile = blockIdx.x * 8 + wid;
  int mt = tile / NT, nt = tile % NT;
  int n = lane & 15, hh = lane >> 4;

  const half_t* arow = ctxh  + (size_t)(mt * 16 + n) * EMB;
  const half_t* brow = wouth + (size_t)(nt * 16 + n) * EMB;

  v8f acc = {};
#pragma unroll
  for (int ks = 0; ks < EMB / 32; ++ks) {
    v16h a  = load_a16x32(arow + ks * 32, hh);
    v16h bm = *(const v16h*)(brow + ks * 32 + hh * 16);
    acc = wmma_f16(a, bm, acc);
  }
  int f = nt * 16 + n;
  float bv = ob[f];
#pragma unroll
  for (int r = 0; r < 8; ++r) {
    int row = mt * 16 + r + 8 * hh;
    attn_out[(size_t)row * EMB + f] = acc[r] + bv;
  }
}

// --------------------------------------------------- residual + layernorm ---
__global__ void ln_kernel(const float* __restrict__ x, const float* __restrict__ attn_out,
                          const float* __restrict__ g, const float* __restrict__ bta,
                          float* __restrict__ out) {
  __shared__ float red[16];
  int row = blockIdx.x;                        // b*SEQ + l
  int b = row / SEQ, l = row % SEQ;
  int e = threadIdx.x;
  float y = x[(size_t)row * EMB + e] + attn_out[((size_t)(b * LP + l)) * EMB + e];
  float s1 = y, s2 = y * y;
#pragma unroll
  for (int off = 1; off < 32; off <<= 1) {
    s1 += __shfl_xor(s1, off, 32);
    s2 += __shfl_xor(s2, off, 32);
  }
  int wid = threadIdx.x >> 5, lane = threadIdx.x & 31;
  if (lane == 0) { red[wid] = s1; red[8 + wid] = s2; }
  __syncthreads();
  float ts1 = 0.f, ts2 = 0.f;
#pragma unroll
  for (int i = 0; i < 8; ++i) { ts1 += red[i]; ts2 += red[8 + i]; }
  float mu  = ts1 * (1.0f / EMB);
  float var = ts2 * (1.0f / EMB) - mu * mu;
  float inv = rsqrtf(var + 1e-5f);
  out[(size_t)row * EMB + e] = (y - mu) * inv * g[e] + bta[e];
}

// ----------------------------------------------------------------- launch ---
extern "C" void kernel_launch(void* const* d_in, const int* in_sizes, int n_in,
                              void* d_out, int out_size, void* d_ws, size_t ws_size,
                              hipStream_t stream) {
  (void)in_sizes; (void)n_in; (void)out_size; (void)ws_size;
  const float* x    = (const float*)d_in[0];
  const float* wqkv = (const float*)d_in[1];
  const float* bq   = (const float*)d_in[2];
  const float* wo   = (const float*)d_in[3];
  const float* bo   = (const float*)d_in[4];
  const float* lng  = (const float*)d_in[5];
  const float* lnb  = (const float*)d_in[6];
  const int* pad_size   = (const int*)d_in[8];
  const int* single_pad = (const int*)d_in[9];

  char* ws = (char*)d_ws;
  size_t off = 0;
  auto alloc = [&](size_t bytes) {
    char* p = ws + off;
    off += (bytes + 255) & ~(size_t)255;
    return p;
  };
  half_t* xh    = (half_t*)alloc((size_t)BATCH * LP * EMB * 2);
  half_t* wqkvh = (half_t*)alloc((size_t)3 * EMB * EMB * 2);
  half_t* wouth = (half_t*)alloc((size_t)EMB * EMB * 2);
  half_t* Qh    = (half_t*)alloc((size_t)BATCH * NH * LP * HD * 2);
  half_t* Kh    = (half_t*)alloc((size_t)BATCH * NH * LP * HD * 2);
  half_t* Vt    = (half_t*)alloc((size_t)BATCH * NH * HD * LP * 2);
  half_t* ctxh  = (half_t*)alloc((size_t)BATCH * LP * EMB * 2);
  float*  aout  = (float*)alloc((size_t)BATCH * LP * EMB * 4);

  prep_x_kernel<<<(BATCH * LP * EMB) / 256, 256, 0, stream>>>(x, xh);
  prep_w_kernel<<<(4 * EMB * EMB) / 256, 256, 0, stream>>>(wqkv, wo, wqkvh, wouth);
  qkv_gemm_kernel<<<((BATCH * LP / 16) * (3 * EMB / 16)) / 8, 256, 0, stream>>>(
      xh, wqkvh, bq, Qh, Kh, Vt);
  attn_kernel<<<BATCH * NH * QBLK, 256, 0, stream>>>(
      Qh, Kh, Vt, pad_size, single_pad, ctxh);
  outproj_gemm_kernel<<<((BATCH * LP / 16) * (EMB / 16)) / 8, 256, 0, stream>>>(
      ctxh, wouth, bo, aout);
  ln_kernel<<<BATCH * SEQ, 256, 0, stream>>>(x, aout, lng, lnb, (float*)d_out);
}
